// HeteroGraphAutoencoder_59742995088081
// MI455X (gfx1250) — matrix operand order repaired and checked by reading, not compiled, not run on hardware
//
#include <hip/hip_runtime.h>

typedef __attribute__((ext_vector_type(16))) _Float16 v16h;
typedef __attribute__((ext_vector_type(8)))  float    v8f;

#define FD   8
#define BINS 32
#define EMB  16
#define HID  64
#define ZD   32

// flags for gemm kernel
#define GF_RELU   1
#define GF_ACCUM  2
#define GF_TRANSW 4

// A-fragment K mapping for v_wmma_f32_16x16x32_f16 (ISA 7.12.2, 16-bit A 16x32):
// lane group g = lane/16; half index i in v16h; VGPR v = i/2, h = i&1
// K = (v<4 ? 2v : 16+2(v-4)) + 8g + h
__device__ __forceinline__ int kmap16(int i, int g) {
  int v = i >> 1, h = i & 1;
  int base = (v < 4) ? (2 * v) : (16 + 2 * (v - 4));
  return base + 8 * g + h;
}

// ---------------------------------------------------------------------------
// Fused embedding gather + pre-MLP:  x0 = relu(embed(x) @ W[128,64] + b)
// One wave per 16-node tile, 8 waves (128 nodes) per block. Weights staged
// into LDS as f16; embedding values gathered straight into the A fragment.
// ---------------------------------------------------------------------------
__global__ void embed_pre_kernel(const int* __restrict__ x,
                                 const float* __restrict__ emb,   // [FD][BINS][EMB]
                                 const float* __restrict__ W,     // [128][64]
                                 const float* __restrict__ bias,  // [64]
                                 float* __restrict__ x0,          // [n][64]
                                 int n) {
  __shared__ _Float16 wl[128 * HID];
  __shared__ float bl[HID];
  const int tid = threadIdx.x;
  for (int i = tid; i < 128 * HID; i += 256) wl[i] = (_Float16)W[i];
  if (tid < HID) bl[tid] = bias[tid];
  __syncthreads();

  const int lane = tid & 31;
  const int wave = tid >> 5;
  const int m0 = (blockIdx.x * 8 + wave) * 16;
  if (m0 >= n) return;                 // uniform per wave -> EXEC stays all-1
  const int g = lane >> 4;
  const int mrow = m0 + (lane & 15);
  const int arow = (mrow < n) ? mrow : (n - 1);

  int bins[FD];
#pragma unroll
  for (int f = 0; f < FD; ++f) bins[f] = x[arow * FD + f];

  v8f acc[4];
#pragma unroll
  for (int t = 0; t < 4; ++t) acc[t] = (v8f){0, 0, 0, 0, 0, 0, 0, 0};

#pragma unroll
  for (int kk = 0; kk < 128; kk += 32) {
    v16h a;
#pragma unroll
    for (int i = 0; i < 16; ++i) {
      int K = kk + kmap16(i, g);
      int f = K >> 4, e = K & 15;
      a[i] = (_Float16)emb[(f * BINS + bins[f]) * EMB + e];
    }
#pragma unroll
    for (int t = 0; t < 4; ++t) {
      v16h bb;
#pragma unroll
      for (int h = 0; h < 16; ++h)
        bb[h] = wl[(kk + kmap16(h, g)) * HID + t * 16 + (lane & 15)];  // lane%16 = N, half = K (kmap16)
      acc[t] = __builtin_amdgcn_wmma_f32_16x16x32_f16(
          false, a, false, bb, (short)0, acc[t], false, false);
    }
  }

#pragma unroll
  for (int t = 0; t < 4; ++t) {
#pragma unroll
    for (int r = 0; r < 8; ++r) {
      int row = m0 + r + 8 * g;
      int col = t * 16 + (lane & 15);
      float v = acc[t][r] + bl[col];
      v = v > 0.f ? v : 0.f;
      if (row < n) x0[row * HID + col] = v;
    }
  }
}

// ---------------------------------------------------------------------------
// Generic WMMA GEMM: D[M,N](+)= scale*(A[M,K] @ W + bias), optional relu,
// optional accumulate into prior D, optional W supplied transposed ([N][K]).
// A is f32 (lda == K), converted to f16 in registers; W staged in LDS as f16.
// Each block: 8 waves x 16 rows = 128 rows, NB columns (blockIdx.y tiles N).
// ---------------------------------------------------------------------------
template <int K, int NB>
__global__ void gemm_kernel(const float* __restrict__ A,
                            const float* __restrict__ W,
                            const float* __restrict__ bias,  // may be null
                            float* __restrict__ D,
                            int M, int N, int flags, float out_scale) {
  __shared__ _Float16 wl[K * NB];
  __shared__ float bl[NB];
  const int tid = threadIdx.x;
  const int c0 = blockIdx.y * NB;
  const bool transw = (flags & GF_TRANSW) != 0;
  for (int i = tid; i < K * NB; i += 256) {
    int k = i / NB, c = i % NB;
    float w = transw ? W[(c0 + c) * K + k] : W[k * N + c0 + c];
    wl[i] = (_Float16)w;
  }
  if (tid < NB) bl[tid] = bias ? bias[c0 + tid] : 0.f;
  __syncthreads();

  const int lane = tid & 31;
  const int wave = tid >> 5;
  const int m0 = (blockIdx.x * 8 + wave) * 16;
  if (m0 >= M) return;
  const int g = lane >> 4;
  const int mrow = m0 + (lane & 15);
  const int arow = (mrow < M) ? mrow : (M - 1);

  constexpr int NT = NB / 16;
  v8f acc[NT];
#pragma unroll
  for (int t = 0; t < NT; ++t) acc[t] = (v8f){0, 0, 0, 0, 0, 0, 0, 0};

#pragma unroll
  for (int kk = 0; kk < K; kk += 32) {
    v16h a;
#pragma unroll
    for (int i = 0; i < 16; ++i)
      a[i] = (_Float16)A[arow * K + kk + kmap16(i, g)];
#pragma unroll
    for (int t = 0; t < NT; ++t) {
      v16h bb;
#pragma unroll
      for (int h = 0; h < 16; ++h)
        bb[h] = wl[(kk + kmap16(h, g)) * NB + t * 16 + (lane & 15)];
      acc[t] = __builtin_amdgcn_wmma_f32_16x16x32_f16(
          false, a, false, bb, (short)0, acc[t], false, false);
    }
  }

#pragma unroll
  for (int t = 0; t < NT; ++t) {
#pragma unroll
    for (int r = 0; r < 8; ++r) {
      int row = m0 + r + 8 * g;
      int col = c0 + t * 16 + (lane & 15);
      float v = acc[t][r] + bl[t * 16 + (lane & 15)];
      if ((flags & GF_ACCUM) && row < M) v += D[row * N + col];
      v *= out_scale;
      if (flags & GF_RELU) v = v > 0.f ? v : 0.f;
      if (row < M) D[row * N + col] = v;
    }
  }
}

// ---------------------------------------------------------------------------
// Edge scatter-add for segment mean: one thread per (edge, column).
// ---------------------------------------------------------------------------
__global__ void scatter_add_kernel(const float* __restrict__ x0,
                                   const int* __restrict__ ei,  // [2][E]
                                   float* __restrict__ m, float* __restrict__ cnt,
                                   int E) {
  long long t = (long long)blockIdx.x * 256 + threadIdx.x;
  if (t >= (long long)E * HID) return;
  int e = (int)(t >> 6);
  int j = (int)(t & 63);
  int s = ei[e];
  int d = ei[E + e];
  atomicAdd(&m[(long long)d * HID + j], x0[(long long)s * HID + j]);
  if (j == 0) atomicAdd(&cnt[d], 1.0f);
}

__global__ void mean_div_kernel(float* __restrict__ m, const float* __restrict__ cnt,
                                int n) {
  int t = blockIdx.x * 256 + threadIdx.x;
  if (t >= n * HID) return;
  float c = cnt[t >> 6];
  m[t] /= (c > 1.f ? c : 1.f);
}

// ---------------------------------------------------------------------------
// Bilinear edge score with factored form: s[e] = dot(z_src[ei0], u[ei1]) + b
// 8 lanes per edge, float4 loads, xor-shuffle reduction within 8-lane group.
// ---------------------------------------------------------------------------
__global__ void bilinear_edge_kernel(const float* __restrict__ zsrc,
                                     const float* __restrict__ u,
                                     const int* __restrict__ ei,  // [2][E]
                                     const float* __restrict__ bias,
                                     float* __restrict__ s, int E) {
  int t = blockIdx.x * 256 + threadIdx.x;
  int e = t >> 3;
  int l8 = t & 7;
  if (e >= E) return;
  int si = ei[e];
  int di = ei[E + e];
  const float4 a = *(const float4*)&zsrc[(long long)si * ZD + l8 * 4];
  const float4 b = *(const float4*)&u[(long long)di * ZD + l8 * 4];
  float p = a.x * b.x + a.y * b.y + a.z * b.z + a.w * b.w;
  p += __shfl_xor(p, 1, 32);
  p += __shfl_xor(p, 2, 32);
  p += __shfl_xor(p, 4, 32);
  if (l8 == 0) s[e] = p + bias[0];
}

// ---------------------------------------------------------------------------
extern "C" void kernel_launch(void* const* d_in, const int* in_sizes, int n_in,
                              void* d_out, int out_size, void* d_ws, size_t ws_size,
                              hipStream_t stream) {
  (void)n_in; (void)out_size; (void)ws_size;
  const int* x_A  = (const int*)d_in[0];
  const int* x_B  = (const int*)d_in[1];
  const int* ei_ab = (const int*)d_in[2];
  const int* ei_ba = (const int*)d_in[3];
  const int* ei_aa = (const int*)d_in[4];
  const float* emb_A = (const float*)d_in[5];
  const float* emb_B = (const float*)d_in[6];
  const float* pre_W_A = (const float*)d_in[7];
  const float* pre_b_A = (const float*)d_in[8];
  const float* pre_W_B = (const float*)d_in[9];
  const float* pre_b_B = (const float*)d_in[10];
  const float* sl_W_ab = (const float*)d_in[11];
  const float* sl_b_ab = (const float*)d_in[12];
  const float* sr_W_ab = (const float*)d_in[13];
  const float* sl_W_ba = (const float*)d_in[14];
  const float* sl_b_ba = (const float*)d_in[15];
  const float* sr_W_ba = (const float*)d_in[16];
  const float* sl_W_aa = (const float*)d_in[17];
  const float* sl_b_aa = (const float*)d_in[18];
  const float* sr_W_aa = (const float*)d_in[19];
  const float* proj_W_A = (const float*)d_in[20];
  const float* proj_b_A = (const float*)d_in[21];
  const float* proj_W_B = (const float*)d_in[22];
  const float* proj_b_B = (const float*)d_in[23];
  const float* head_W_A = (const float*)d_in[24];
  const float* head_b_A = (const float*)d_in[25];
  const float* head_W_B = (const float*)d_in[26];
  const float* head_b_B = (const float*)d_in[27];
  const float* bil_W_ab = (const float*)d_in[28];
  const float* bil_b_ab = (const float*)d_in[29];
  const float* bil_W_ba = (const float*)d_in[30];
  const float* bil_b_ba = (const float*)d_in[31];
  const float* bil_W_aa = (const float*)d_in[32];
  const float* bil_b_aa = (const float*)d_in[33];

  const int N = in_sizes[0] / FD;       // 50000 (N_A == N_B)
  const int E = in_sizes[2] / 2;        // 800000

  // workspace layout (floats); mB..cA2 form one contiguous zeroed region
  float* ws = (float*)d_ws;
  float* x0A = ws;                       // N*64
  float* x0B = x0A + (size_t)N * HID;    // N*64
  float* mB  = x0B + (size_t)N * HID;    // N*64  (zeroed)
  float* mA1 = mB  + (size_t)N * HID;    // N*64  (zeroed)
  float* mA2 = mA1 + (size_t)N * HID;    // N*64  (zeroed)
  float* cB  = mA2 + (size_t)N * HID;    // N     (zeroed)
  float* cA1 = cB  + (size_t)N;          // N     (zeroed)
  float* cA2 = cA1 + (size_t)N;          // N     (zeroed)
  float* hA  = cA2 + (size_t)N;          // N*64
  float* hB  = hA  + (size_t)N * HID;    // N*64
  float* u   = hB  + (size_t)N * HID;    // N*32 (reused for 3 bilinears)

  // output layout (floats, reference return order)
  float* zA  = (float*)d_out;            // N*32
  float* zB  = zA  + (size_t)N * ZD;     // N*32
  float* flA = zB  + (size_t)N * ZD;     // N*256
  float* flB = flA + (size_t)N * FD * BINS;
  float* sab = flB + (size_t)N * FD * BINS;
  float* sba = sab + (size_t)E;
  float* saa = sba + (size_t)E;

  const int rowBlocks = (N + 127) / 128;

  // zero the segment-sum accumulators (contiguous region mB..cA2)
  (void)hipMemsetAsync(mB, 0, ((size_t)3 * N * HID + 3 * N) * sizeof(float), stream);

  // 1) embed + pre-MLP (WMMA)
  embed_pre_kernel<<<rowBlocks, 256, 0, stream>>>(x_A, emb_A, pre_W_A, pre_b_A, x0A, N);
  embed_pre_kernel<<<rowBlocks, 256, 0, stream>>>(x_B, emb_B, pre_W_B, pre_b_B, x0B, N);

  // 2) segment means over edges
  {
    long long work = (long long)E * HID;
    int blks = (int)((work + 255) / 256);
    scatter_add_kernel<<<blks, 256, 0, stream>>>(x0A, ei_ab, mB,  cB,  E);
    scatter_add_kernel<<<blks, 256, 0, stream>>>(x0B, ei_ba, mA1, cA1, E);
    scatter_add_kernel<<<blks, 256, 0, stream>>>(x0A, ei_aa, mA2, cA2, E);
  }
  {
    int blks = (N * HID + 255) / 256;
    mean_div_kernel<<<blks, 256, 0, stream>>>(mB,  cB,  N);
    mean_div_kernel<<<blks, 256, 0, stream>>>(mA1, cA1, N);
    mean_div_kernel<<<blks, 256, 0, stream>>>(mA2, cA2, N);
  }

  // 3) SAGE dense parts (WMMA):
  // h_B = mB@Wl_ab + bl_ab + x0B@Wr_ab
  gemm_kernel<HID, 64><<<dim3(rowBlocks, 1), 256, 0, stream>>>(mB,  sl_W_ab, sl_b_ab, hB, N, HID, 0, 1.f);
  gemm_kernel<HID, 64><<<dim3(rowBlocks, 1), 256, 0, stream>>>(x0B, sr_W_ab, nullptr, hB, N, HID, GF_ACCUM, 1.f);
  // h_A = 0.5*(mA1@Wl_ba + bl_ba + x0A@Wr_ba + mA2@Wl_aa + bl_aa + x0A@Wr_aa)
  gemm_kernel<HID, 64><<<dim3(rowBlocks, 1), 256, 0, stream>>>(mA1, sl_W_ba, sl_b_ba, hA, N, HID, 0, 1.f);
  gemm_kernel<HID, 64><<<dim3(rowBlocks, 1), 256, 0, stream>>>(x0A, sr_W_ba, nullptr, hA, N, HID, GF_ACCUM, 1.f);
  gemm_kernel<HID, 64><<<dim3(rowBlocks, 1), 256, 0, stream>>>(mA2, sl_W_aa, sl_b_aa, hA, N, HID, GF_ACCUM, 1.f);
  gemm_kernel<HID, 64><<<dim3(rowBlocks, 1), 256, 0, stream>>>(x0A, sr_W_aa, nullptr, hA, N, HID, GF_ACCUM, 0.5f);

  // 4) projection to z (relu)
  gemm_kernel<HID, 32><<<dim3(rowBlocks, 1), 256, 0, stream>>>(hA, proj_W_A, proj_b_A, zA, N, ZD, GF_RELU, 1.f);
  gemm_kernel<HID, 32><<<dim3(rowBlocks, 1), 256, 0, stream>>>(hB, proj_W_B, proj_b_B, zB, N, ZD, GF_RELU, 1.f);

  // 5) decoder heads fl = z @ headW + b  (N=256 split into 4 column blocks)
  gemm_kernel<ZD, 64><<<dim3(rowBlocks, 4), 256, 0, stream>>>(zA, head_W_A, head_b_A, flA, N, FD * BINS, 0, 1.f);
  gemm_kernel<ZD, 64><<<dim3(rowBlocks, 4), 256, 0, stream>>>(zB, head_W_B, head_b_B, flB, N, FD * BINS, 0, 1.f);

  // 6) bilinear edge scores, factored: u = z_dst @ W^T, s = dot(z_src, u) + b
  {
    int eblks = (E * 8 + 255) / 256;
    gemm_kernel<ZD, 32><<<dim3(rowBlocks, 1), 256, 0, stream>>>(zB, bil_W_ab, nullptr, u, N, ZD, GF_TRANSW, 1.f);
    bilinear_edge_kernel<<<eblks, 256, 0, stream>>>(zA, u, ei_ab, bil_b_ab, sab, E);
    gemm_kernel<ZD, 32><<<dim3(rowBlocks, 1), 256, 0, stream>>>(zA, bil_W_ba, nullptr, u, N, ZD, GF_TRANSW, 1.f);
    bilinear_edge_kernel<<<eblks, 256, 0, stream>>>(zB, u, ei_ba, bil_b_ba, sba, E);
    gemm_kernel<ZD, 32><<<dim3(rowBlocks, 1), 256, 0, stream>>>(zA, bil_W_aa, nullptr, u, N, ZD, GF_TRANSW, 1.f);
    bilinear_edge_kernel<<<eblks, 256, 0, stream>>>(zA, u, ei_aa, bil_b_aa, saa, E);
  }
}
